// LogSparceAttention_26963804684336
// MI455X (gfx1250) — matrix operand order repaired and checked by reading, not compiled, not run on hardware
//
#include <hip/hip_runtime.h>
#include <math.h>

typedef __attribute__((ext_vector_type(16))) _Float16 v16h;
typedef __attribute__((ext_vector_type(8)))  float    v8f;

namespace {
constexpr int Bc = 2;
constexpr int Lc = 2048;
constexpr int Hc = 16;
constexpr int Ec = 64;
constexpr int QT = Lc / 16;          // 128 query tiles per (b,h)
constexpr float SCALE = 0.125f;      // 1/sqrt(E), E = 64
// log-sparse mask parameters for win_len = sub_len = 2048 (log_l = 11):
constexpr int THRESH = 22;           // (win_len//sub_len)*2*log_l
constexpr int BAND   = 10;           // log_l - 1
constexpr int DMAX   = 1024;         // 2^(log_l-1)
}

// Exact per-element port of LogSparceAttention.row_mask for these shapes.
__device__ __forceinline__ bool elem_allowed(int i, int j) {
  if (j > i) return false;
  if (i < THRESH) return true;                 // full causal rows
  if (i - j <= BAND) return true;              // diagonal band [i-10, i]
  int d = i - BAND - j;                        // allowed iff d == 2^p, p in [0,10]
  return (d >= 1) && (d <= DMAX) && ((d & (d - 1)) == 0);
}

// Does k-tile t contain any unmasked element for query tile qt? (uniform)
__device__ __forceinline__ bool tile_needed(int qt, int t) {
  if (t >= qt - 1) return true;                // diagonal band tiles
  if (qt <= 1) return true;                    // rows i < 22 are full causal
  int delta = 16 * (qt - t);                   // d range hit by this tile pair
  int lo = delta - 25; if (lo < 1) lo = 1;
  int hi = delta + 5;  if (hi > DMAX) hi = DMAX;
#pragma unroll
  for (int p = 0; p <= 10; ++p) {
    int d = 1 << p;
    if (d >= lo && d <= hi) return true;
  }
  return false;
}

__global__ __launch_bounds__(32) void logsparse_attn_wmma(
    const float* __restrict__ Q, const float* __restrict__ K,
    const float* __restrict__ V, float* __restrict__ Out) {
  const int lane = threadIdx.x;                // wave32
  const int li   = lane & 15;
  const int half = lane >> 4;

  const int bid = blockIdx.x;
  const int qt = bid % QT;
  const int h  = (bid / QT) % Hc;
  const int b  = bid / (QT * Hc);
  const int i0 = qt * 16;
  const int irow = i0 + li;

  const size_t HE = (size_t)Hc * Ec;           // stride between sequence rows

  // ---- Q as B-operand of GEMM1 (S^T = K_tile x Q^T) ----
  // B operand (16x16x32), contiguous k order chosen here: lane%16 = N (= i_local); elem e -> Kdim = half*16 + e
  const float* qrow = Q + (((size_t)b * Lc + irow) * Hc + h) * Ec;
  v16h qb0, qb1;
#pragma unroll
  for (int e = 0; e < 16; ++e) {
    qb0[e] = (_Float16)qrow[half * 16 + e];         // e-dim 0..31
    qb1[e] = (_Float16)qrow[32 + half * 16 + e];    // e-dim 32..63
  }

  v8f oacc[4] = {};                            // O tile: lane%16 = e_out, elem r -> row r+8*half
  float mrow = -1e30f;                         // running max for row (i0 + li)
  float lrow = 0.0f;                           // running sum for row (i0 + li)

  for (int t = 0; t <= qt; ++t) {              // ascending: diagonal tile last
    if (!tile_needed(qt, t)) continue;

    // ---- K tile as A-operand: lane%16 = M (= j_local)
    // A operand, contiguous k order chosen here: elem e -> K = half*16 + e (matches Q B-operand)
    const float* krow = K + (((size_t)b * Lc + (16 * t + li)) * Hc + h) * Ec;
    v16h ka0, ka1;
#pragma unroll
    for (int e = 0; e < 8; ++e) {
      ka0[e]     = (_Float16)krow[half * 16 + e];
      ka0[e + 8] = (_Float16)krow[half * 16 + 8 + e];
      ka1[e]     = (_Float16)krow[32 + half * 16 + e];
      ka1[e + 8] = (_Float16)krow[40 + half * 16 + e];
    }

    v8f st = {};
    st = __builtin_amdgcn_wmma_f32_16x16x32_f16(false, ka0, false, qb0,
                                                (short)0, st, false, false);
    st = __builtin_amdgcn_wmma_f32_16x16x32_f16(false, ka1, false, qb1,
                                                (short)0, st, false, false);
    // st = S^T tile: lane holds query row i = irow; elem r -> key j = 16t + 8*half + r

    float sc[8];
#pragma unroll
    for (int r = 0; r < 8; ++r) {
      int j = 16 * t + 8 * half + r;
      sc[r] = elem_allowed(irow, j) ? st[r] * SCALE : -1e30f;
    }

    // online softmax: row lives in this lane's 8 elems + partner lane (lane^16)
    float tmax = sc[0];
#pragma unroll
    for (int r = 1; r < 8; ++r) tmax = fmaxf(tmax, sc[r]);
    tmax = fmaxf(tmax, __shfl_xor(tmax, 16, 32));
    float mnew = fmaxf(mrow, tmax);
    float corr = __expf(mrow - mnew);

    float ps[8];
    float psum = 0.0f;
#pragma unroll
    for (int r = 0; r < 8; ++r) { ps[r] = __expf(sc[r] - mnew); psum += ps[r]; }
    psum += __shfl_xor(psum, 16, 32);
    lrow = lrow * corr + psum;
    mrow = mnew;

    // rescale O: element r belongs to output row (r + 8*half); corr for row k
    // is held by lane k (and k+16) -> broadcast
    float cr[8];
#pragma unroll
    for (int r = 0; r < 8; ++r) cr[r] = __shfl(corr, r + 8 * half, 32);
#pragma unroll
    for (int n = 0; n < 4; ++n)
#pragma unroll
      for (int r = 0; r < 8; ++r) oacc[n][r] *= cr[r];

    // P tile drops straight into the A-operand layout of GEMM2 (no transpose):
    // elem e (0..7) -> K = j_local = half*8 + e == our ps[e]; K = 16..31 zero-padded
    v16h pa;
#pragma unroll
    for (int e = 0; e < 8; ++e) { pa[e] = (_Float16)ps[e]; pa[e + 8] = (_Float16)0.0f; }

    // V tile as B-operand, 4 output-column tiles of 16.
    // elem e<8 -> K = j_local = half*8 + e; elems 8..15 multiply the zero half
    // of A, so they may load any (finite) rows of the tile.
    const float* vbase = V + (((size_t)b * Lc + 16 * t) * Hc + h) * Ec + li;
#pragma unroll
    for (int n = 0; n < 4; ++n) {
      v16h vb;
      const float* vcol = vbase + 16 * n;
#pragma unroll
      for (int e = 0; e < 16; ++e) vb[e] = (_Float16)vcol[(size_t)((half * 8 + e) & 15) * HE];
      oacc[n] = __builtin_amdgcn_wmma_f32_16x16x32_f16(false, pa, false, vb,
                                                       (short)0, oacc[n], false, false);
    }
  }

  // final 1/l normalization (per output row, redistributed by shuffle) + store
  float linv[8];
#pragma unroll
  for (int r = 0; r < 8; ++r) {
    float lr = __shfl(lrow, r + 8 * half, 32);
    linv[r] = 1.0f / lr;                       // every row has its diagonal -> l > 0
  }
#pragma unroll
  for (int n = 0; n < 4; ++n) {
#pragma unroll
    for (int r = 0; r < 8; ++r) {
      int orow = i0 + r + 8 * half;
      Out[(((size_t)b * Lc + orow) * Hc + h) * Ec + 16 * n + li] =
          oacc[n][r] * linv[r];
    }
  }
}

extern "C" void kernel_launch(void* const* d_in, const int* in_sizes, int n_in,
                              void* d_out, int out_size, void* d_ws, size_t ws_size,
                              hipStream_t stream) {
  const float* q = (const float*)d_in[0];
  const float* k = (const float*)d_in[1];
  const float* v = (const float*)d_in[2];
  float* out = (float*)d_out;
  (void)in_sizes; (void)n_in; (void)out_size; (void)d_ws; (void)ws_size;

  dim3 grid(Bc * Hc * QT);   // 2 * 16 * 128 = 4096 waves, one 16-query tile each
  dim3 block(32);            // one wave32
  hipLaunchKernelGGL(logsparse_attn_wmma, grid, block, 0, stream, q, k, v, out);
}
